// StrideOptimizedGroupedLinear_44195213476070
// MI455X (gfx1250) — matrix operand order repaired and checked by reading, not compiled, not run on hardware
//
#include <hip/hip_runtime.h>

typedef __attribute__((ext_vector_type(16))) __bf16 v16bf;
typedef __attribute__((ext_vector_type(8)))  __bf16 v8bf;
typedef __attribute__((ext_vector_type(8)))  float  v8f;

#define IN_F   2048
#define OUT_F  4096
#define BM     256         // block tile rows (4 waves x 64)
#define BN     128         // block tile cols (2 waves x 64)
#define BK     32          // one bf16-WMMA K step
#define LDK    40          // BK + 8 bf16 pad: 80B row stride, 16B-aligned, spreads banks
#define NTHR   256         // 8 wave32 waves

struct Staged { v8bf a[4]; v8bf b[2]; };

__global__ __launch_bounds__(NTHR) void moe_grouped_gemm_bf16_kernel(
    const float* __restrict__ A,        // [T, IN_F] fp32 tokens (pre-sorted by expert)
    const float* __restrict__ W,        // [E, OUT_F, IN_F] fp32 weights
    const int*   __restrict__ m_sizes,  // [E]
    const int*   __restrict__ m_offsets,// [E]
    float*       __restrict__ out,      // [T, OUT_F]
    int E_num)
{
  __shared__ __bf16 As[2][BM][LDK];    // 40 KB
  __shared__ __bf16 Bs[2][BN][LDK];    // 20 KB

  const int tid  = threadIdx.x;
  const int lane = tid & 31;
  const int wave = tid >> 5;
  const int wm   = wave & 3;    // 4 waves along M, 64 rows each
  const int wn   = wave >> 2;   // 2 waves along N, 64 cols each

  // ---- map blockIdx.y -> (expert, m-tile) over ragged groups (E is tiny) ----
  int e = -1, mtile = 0;
  {
    int y = blockIdx.y, acc = 0;
    for (int i = 0; i < E_num; ++i) {
      int nt = (m_sizes[i] + BM - 1) / BM;
      if (y < acc + nt) { e = i; mtile = y - acc; break; }
      acc += nt;
    }
  }
  if (e < 0) return;   // block-uniform exit: safe w.r.t. barriers / EXEC

  const int m_off = m_offsets[e];
  const int m_hi  = m_off + m_sizes[e];       // exclusive row bound for this expert
  const int m0    = m_off + mtile * BM;
  const int n0    = blockIdx.x * BN;

  const float* Wg = W + (long long)e * OUT_F * IN_F;

  // ---- staging: each thread moves 4 A chunks + 2 B chunks of 8 elems per K-step ----
  const int rr = tid >> 2;          // 0..63
  const int kc = (tid & 3) * 8;     // k chunk within BK: 0,8,16,24

  const float* pa[4];
  #pragma unroll
  for (int j = 0; j < 4; ++j) {
    int ga = m0 + rr + 64 * j;
    if (ga >= m_hi) ga = m_hi - 1;            // clamp ragged edge (store is masked)
    pa[j] = A + (long long)ga * IN_F + kc;
  }
  const float* pb[2];
  #pragma unroll
  for (int j = 0; j < 2; ++j)
    pb[j] = Wg + (long long)(n0 + rr + 64 * j) * IN_F + kc;

  auto fetch = [&](int k0) -> Staged {
    Staged s;
    #pragma unroll
    for (int j = 0; j < 4; ++j)
      s.a[j] = __builtin_convertvector(*(const v8f*)(pa[j] + k0), v8bf); // fp32->bf16 RNE
    #pragma unroll
    for (int j = 0; j < 2; ++j)
      s.b[j] = __builtin_convertvector(*(const v8f*)(pb[j] + k0), v8bf);
    if (k0 + BK < IN_F) {           // speculative prefetch of the tile after next
      __builtin_prefetch(pa[0] + k0 + BK, 0, 0);
      __builtin_prefetch(pb[0] + k0 + BK, 0, 0);
    }
    return s;
  };

  auto stage = [&](const Staged& s, int buf) {
    #pragma unroll
    for (int j = 0; j < 4; ++j)
      *(v8bf*)(&As[buf][rr + 64 * j][kc]) = s.a[j];
    #pragma unroll
    for (int j = 0; j < 2; ++j)
      *(v8bf*)(&Bs[buf][rr + 64 * j][kc]) = s.b[j];
  };

  v8f acc[4][4];
  #pragma unroll
  for (int i = 0; i < 4; ++i)
    #pragma unroll
    for (int j = 0; j < 4; ++j) acc[i][j] = 0.0f;

  const int frow = lane & 15;   // fragment row/col within 16
  const int fhi  = lane >> 4;   // half-wave select

  Staged s = fetch(0);
  stage(s, 0);
  __syncthreads();

  const int NK = IN_F / BK;
  for (int kt = 0; kt < NK; ++kt) {
    const int  buf  = kt & 1;
    const bool more = (kt + 1) < NK;
    if (more) s = fetch((kt + 1) * BK);

    // Load ALL fragments first, then a 16-deep dependence-free WMMA burst.
    // A fragments: 16x32 bf16 (ISA 7.12.2): lane holds K {kb..kb+7, kb+16..kb+23}, kb = 8*fhi
    v16bf af[4];
    #pragma unroll
    for (int tm = 0; tm < 4; ++tm) {
      const __bf16* row = &As[buf][wm * 64 + tm * 16 + frow][fhi * 8];
      v8bf lo = *(const v8bf*)(row);
      v8bf hi = *(const v8bf*)(row + 16);
      af[tm] = __builtin_shufflevector(lo, hi, 0,1,2,3,4,5,6,7,8,9,10,11,12,13,14,15);
    }
    // B fragments: 32x16 bf16: lane holds K {kb..kb+7, kb+16..kb+23}, kb = 8*fhi, N = frow.
    // W rows are [n][k] (K contiguous) => direct load, stride-based transpose avoidance.
    v16bf bfr[4];
    #pragma unroll
    for (int tn = 0; tn < 4; ++tn) {
      const __bf16* col = &Bs[buf][wn * 64 + tn * 16 + frow][fhi * 8];
      v8bf lo = *(const v8bf*)(col);
      v8bf hi = *(const v8bf*)(col + 16);
      bfr[tn] = __builtin_shufflevector(lo, hi, 0,1,2,3,4,5,6,7,8,9,10,11,12,13,14,15);
    }
    #pragma unroll
    for (int tn = 0; tn < 4; ++tn)
      #pragma unroll
      for (int tm = 0; tm < 4; ++tm)
        acc[tm][tn] = __builtin_amdgcn_wmma_f32_16x16x32_bf16(
            /*neg_a=*/false, af[tm], /*neg_b=*/false, bfr[tn],
            /*c_mod=*/(short)0, acc[tm][tn], /*reuse_a=*/false, /*reuse_b=*/false);

    if (more) stage(s, buf ^ 1);   // write other buffer; one barrier/iter is sufficient
    __syncthreads();
  }

  // ---- write back 256x128 fp32 tile; C/D layout: elem r -> M = r + 8*fhi, N = frow ----
  #pragma unroll
  for (int tm = 0; tm < 4; ++tm) {
    const int rbase = m0 + wm * 64 + tm * 16 + fhi * 8;
    #pragma unroll
    for (int tn = 0; tn < 4; ++tn) {
      const int c = n0 + wn * 64 + tn * 16 + frow;
      #pragma unroll
      for (int r = 0; r < 8; ++r) {
        const int row = rbase + r;
        if (row < m_hi) out[(long long)row * OUT_F + c] = acc[tm][tn][r];
      }
    }
  }
}

extern "C" void kernel_launch(void* const* d_in, const int* in_sizes, int n_in,
                              void* d_out, int out_size, void* d_ws, size_t ws_size,
                              hipStream_t stream) {
  const float* tokens    = (const float*)d_in[0];
  const float* weights   = (const float*)d_in[1];
  const int*   m_sizes   = (const int*)d_in[2];
  const int*   m_offsets = (const int*)d_in[3];
  float*       out       = (float*)d_out;

  const int E_num = in_sizes[2];
  const int T     = in_sizes[0] / IN_F;

  // grid.y upper-bounds sum(ceil(m_e/BM)) <= T/BM + E; out-of-range blocks exit early.
  dim3 grid(OUT_F / BN, T / BM + E_num);
  moe_grouped_gemm_bf16_kernel<<<grid, NTHR, 0, stream>>>(
      tokens, weights, m_sizes, m_offsets, out, E_num);
  (void)d_ws; (void)ws_size; (void)n_in; (void)out_size;
}
